// SSMCardiacGNN_62268435857643
// MI455X (gfx1250) — compile-verified, matrix operand order checked by reading, not run on hardware
//
#include <hip/hip_runtime.h>
#include <hip/hip_bf16.h>
#include <math.h>

// ---- static problem sizes ----
#define NNODES 16384
#define NPGR   4096
#define NBAT   4
#define NEDGES 262144
#define NE1    (NEDGES + NNODES)   // with self loops
#define DH     128
#define NHEAD  2
#define KD     16
#define KPCA   25
#define NEGS   0.2f
#define EPSB   1e-5f

typedef float  v2f   __attribute__((ext_vector_type(2)));
typedef float  v8f   __attribute__((ext_vector_type(8)));
typedef __bf16 v16bf __attribute__((ext_vector_type(16)));
typedef __bf16 bf16x8 __attribute__((ext_vector_type(8)));

// ---------------- small utilities ----------------
__global__ void k_fill(float* p, float v, int n) {
    int t = blockIdx.x * blockDim.x + threadIdx.x;
    if (t < n) p[t] = v;
}

__device__ inline float lrelu(float v) { return v >= 0.f ? v : NEGS * v; }

__device__ inline void atomicMaxF(float* addr, float val) {
    int* ia = (int*)addr;
    int old = __float_as_int(*addr);
    while (__int_as_float(old) < val) {
        int prev = atomicCAS(ia, old, __float_as_int(val));
        if (prev == old) break;
        old = prev;
    }
}

__device__ inline void edge_sd(const int* ei, int e, int& s, int& d) {
    if (e < NEDGES) { s = ei[e]; d = ei[NEDGES + e]; }
    else            { s = e - NEDGES; d = s; }
}

// ---------------- self-loop edge-attr mean ----------------
__global__ void k_selfloop_sum(const int* ei, const float* ea, float* easum, float* cnt) {
    int e = blockIdx.x * blockDim.x + threadIdx.x;
    if (e >= NEDGES) return;
    int d = ei[NEDGES + e];
    atomicAdd(&cnt[d], 1.0f);
#pragma unroll
    for (int j = 0; j < 6; ++j) atomicAdd(&easum[d * 6 + j], ea[e * 6 + j]);
}
__global__ void k_selfloop_div(float* easum, const float* cnt) {
    int t = blockIdx.x * blockDim.x + threadIdx.x;
    if (t >= NNODES * 6) return;
    easum[t] /= fmaxf(cnt[t / 6], 1.0f);
}

// ---------------- K=6 input transform (tiny, VALU) ----------------
__global__ void k_lin6(const float* x, const float* W, const float* b, float* y) {
    int t = blockIdx.x * blockDim.x + threadIdx.x;
    if (t >= NNODES * 256) return;
    int n = t >> 8, o = t & 255;
    float acc = b[o];
#pragma unroll
    for (int j = 0; j < 6; ++j) acc += x[n * 6 + j] * W[j * 256 + o];
    y[t] = acc;
}

// ---------------- generic fp32 WMMA GEMM: C = act(A[MxK] @ B[KxN] + bias) ----------------
// one wave computes a 16x16 tile via v_wmma_f32_16x16x4_f32
__global__ void k_gemm_f32(const float* __restrict__ A, const float* __restrict__ B,
                           const float* __restrict__ bias, float* __restrict__ C,
                           int M, int N, int K, int act) {
    int wave = (blockIdx.x * blockDim.x + threadIdx.x) >> 5;
    int lane = threadIdx.x & 31;
    int tn = N >> 4;
    int ntiles = (M >> 4) * tn;
    if (wave >= ntiles) return;          // wave-uniform
    int m0 = (wave / tn) * 16, n0 = (wave % tn) * 16;
    int r = lane & 15, hi = lane >> 4;
    v8f c = {};
    for (int k4 = 0; k4 < K; k4 += 4) {
        v2f a = *(const v2f*)(A + (size_t)(m0 + r) * K + k4 + 2 * hi);
        v2f b;
        b.x = B[(size_t)(k4 + 2 * hi)     * N + n0 + r];
        b.y = B[(size_t)(k4 + 2 * hi + 1) * N + n0 + r];
        c = __builtin_amdgcn_wmma_f32_16x16x4_f32(false, a, false, b, (short)0, c, false, false);
    }
    float bv = bias ? bias[n0 + r] : 0.f;
#pragma unroll
    for (int j = 0; j < 8; ++j) {
        float v = c[j] + bv;
        if (act == 1) v = lrelu(v);
        else if (act == 2) v = tanhf(v);
        C[(size_t)(m0 + j + 8 * hi) * N + n0 + r] = v;
    }
}

// ---------------- GATv2 edge kernels ----------------
__global__ void k_gat_score(const int* ei, const float* xl, const float* xr,
                            const float* ea, const float* ealoop, const float* We,
                            const float* att, float* s_out, float* m_nh) {
    int t = blockIdx.x * blockDim.x + threadIdx.x;
    if (t >= NE1 * NHEAD) return;
    int e = t >> 1, h = t & 1;
    int s, d; edge_sd(ei, e, s, d);
    float eav[6];
    bool has_ea = (We != nullptr);
    if (has_ea) {
        const float* ep = (e < NEDGES) ? (ea + (size_t)e * 6) : (ealoop + (size_t)(e - NEDGES) * 6);
#pragma unroll
        for (int j = 0; j < 6; ++j) eav[j] = ep[j];
    }
    const float* pl = xl + (size_t)s * 256 + h * DH;
    const float* pr = xr + (size_t)d * 256 + h * DH;
    const float* pa = att + h * DH;
    float acc = 0.f;
    for (int dd = 0; dd < DH; ++dd) {
        float v = pl[dd] + pr[dd];
        if (has_ea) {
            const float* w = We + h * DH + dd;
            float s6 = 0.f;
#pragma unroll
            for (int j = 0; j < 6; ++j) s6 += eav[j] * w[j * 256];
            v += s6;
        }
        acc += lrelu(v) * pa[dd];
    }
    s_out[t] = acc;
    atomicMaxF(&m_nh[d * NHEAD + h], acc);
}

__global__ void k_gat_exp(const int* ei, float* s_io, const float* m_nh, float* den_nh) {
    int t = blockIdx.x * blockDim.x + threadIdx.x;
    if (t >= NE1 * NHEAD) return;
    int e = t >> 1, h = t & 1;
    int s, d; edge_sd(ei, e, s, d);
    float ex = expf(s_io[t] - m_nh[d * NHEAD + h]);
    s_io[t] = ex;
    atomicAdd(&den_nh[d * NHEAD + h], ex);
}

__global__ void k_gat_accum(const int* ei, const float* s_ex, const float* den_nh,
                            const float* xl, float* acc) {
    int t = blockIdx.x * blockDim.x + threadIdx.x;
    if (t >= NE1 * NHEAD) return;
    int e = t >> 1, h = t & 1;
    int s, d; edge_sd(ei, e, s, d);
    float w = s_ex[t] / fmaxf(den_nh[d * NHEAD + h], 1e-16f);
    const float* pl = xl + (size_t)s * 256 + h * DH;
    float* pd = acc + (size_t)d * 256 + h * DH;
    for (int dd = 0; dd < DH; ++dd) atomicAdd(&pd[dd], pl[dd] * w);
}

__global__ void k_headmean_bias(const float* acc, const float* bias, float* h) {
    int t = blockIdx.x * blockDim.x + threadIdx.x;
    if (t >= NNODES * DH) return;
    int n = t >> 7, dd = t & 127;
    h[t] = 0.5f * (acc[(size_t)n * 256 + dd] + acc[(size_t)n * 256 + DH + dd]) + bias[dd];
}

// ---------------- BatchNorm ----------------
__global__ void k_bn_stats(const float* x, float* stat) {
    int d = blockIdx.x, tid = threadIdx.x;
    float s1 = 0.f, s2 = 0.f;
    for (int n = tid; n < NNODES; n += 256) {
        float v = x[(size_t)n * DH + d];
        s1 += v; s2 += v * v;
    }
    __shared__ float r1[256], r2[256];
    r1[tid] = s1; r2[tid] = s2; __syncthreads();
    for (int o = 128; o > 0; o >>= 1) {
        if (tid < o) { r1[tid] += r1[tid + o]; r2[tid] += r2[tid + o]; }
        __syncthreads();
    }
    if (tid == 0) {
        float mu = r1[0] / (float)NNODES;
        float var = r2[0] / (float)NNODES - mu * mu;
        stat[d * 2] = mu; stat[d * 2 + 1] = rsqrtf(var + EPSB);
    }
}
__global__ void k_bn_apply(float* x, const float* stat, const float* g, const float* b, int do_lrelu) {
    int t = blockIdx.x * blockDim.x + threadIdx.x;
    if (t >= NNODES * DH) return;
    int d = t & 127;
    float v = (x[t] - stat[d * 2]) * stat[d * 2 + 1] * g[d] + b[d];
    if (do_lrelu) v = lrelu(v);
    x[t] = v;
}

// ---------------- KNN: bf16 WMMA Gram + top-k ----------------
__global__ void k_sqnorm(const float* h, float* sq) {
    int n = blockIdx.x * blockDim.x + threadIdx.x;
    if (n >= NNODES) return;
    float s = 0.f;
    for (int d = 0; d < DH; ++d) { float v = h[(size_t)n * DH + d]; s += v * v; }
    sq[n] = s;
}
__global__ void k_cvt_bf16(const float* h, __bf16* hb) {
    int t = blockIdx.x * blockDim.x + threadIdx.x;
    if (t >= NNODES * DH) return;
    hb[t] = (__bf16)h[t];
}

__device__ inline v16bf ld_bf16_frag(const __bf16* row, int k, int hi) {
    bf16x8 lo = *(const bf16x8*)(row + k + hi * 8);
    bf16x8 hh = *(const bf16x8*)(row + k + 16 + hi * 8);
    v16bf a;
#pragma unroll
    for (int i = 0; i < 8; ++i) { a[i] = lo[i]; a[8 + i] = hh[i]; }
    return a;
}

__global__ void k_gram(const __bf16* __restrict__ hb, float* __restrict__ G, int graph) {
    int wave = blockIdx.x * 8 + (threadIdx.x >> 5);   // 65536 waves, exact
    int lane = threadIdx.x & 31;
    int ti = wave >> 8, tj = wave & 255;
    const __bf16* base = hb + (size_t)graph * NPGR * DH;
    int r = lane & 15, hi = lane >> 4;
    int i0 = ti * 16, j0 = tj * 16;
    const __bf16* rowA = base + (size_t)(i0 + r) * DH;
    const __bf16* rowB = base + (size_t)(j0 + r) * DH;
    v8f c = {};
    for (int k = 0; k < DH; k += 32) {
        v16bf a = ld_bf16_frag(rowA, k, hi);
        v16bf b = ld_bf16_frag(rowB, k, hi);
        c = __builtin_amdgcn_wmma_f32_16x16x32_bf16(false, a, false, b, (short)0, c, false, false);
    }
#pragma unroll
    for (int j = 0; j < 8; ++j)
        G[(size_t)(i0 + j + 8 * hi) * NPGR + (j0 + r)] = c[j];
}

__global__ void k_topk(const float* __restrict__ G, const float* __restrict__ sq,
                       int* __restrict__ col, int graph) {
    int i = blockIdx.x * blockDim.x + threadIdx.x;
    if (i >= NPGR) return;
    int gb = graph * NPGR;
    float sqi = sq[gb + i];
    const float* row = G + (size_t)i * NPGR;
    float bd[KD]; int bi[KD];
#pragma unroll
    for (int k = 0; k < KD; ++k) { bd[k] = 3.4e38f; bi[k] = 0; }
    for (int j = 0; j < NPGR; ++j) {
        if (j == i) continue;
        float d2 = sqi + sq[gb + j] - 2.f * row[j];
        if (d2 < bd[KD - 1]) {
            bd[KD - 1] = d2; bi[KD - 1] = j;
#pragma unroll
            for (int t = KD - 1; t > 0; --t) {
                if (bd[t] < bd[t - 1]) {
                    float tf = bd[t]; bd[t] = bd[t - 1]; bd[t - 1] = tf;
                    int   ti2 = bi[t]; bi[t] = bi[t - 1]; bi[t - 1] = ti2;
                }
            }
        }
    }
#pragma unroll
    for (int k = 0; k < KD; ++k) col[(size_t)(gb + i) * KD + k] = gb + bi[k];
}

// ---------------- fused edge-MLP conv (fp32 WMMA) ----------------
// one wave handles one node's 16 messages end-to-end; 2 waves / block
#define EMW 2
__global__ __launch_bounds__(64) void k_edge_mlp(const float* __restrict__ h,
        const int* __restrict__ col,
        const float* __restrict__ W1, const float* __restrict__ b1,
        const float* __restrict__ W2, const float* __restrict__ b2,
        float* __restrict__ out) {
    __shared__ float cat[EMW][16][256];
    __shared__ float tb[EMW][16][128];
    int lane = threadIdx.x & 31, wv = threadIdx.x >> 5;
    int n = blockIdx.x * EMW + wv;
    for (int idx = lane; idx < 16 * 256; idx += 32) {
        int r = idx >> 8, j = idx & 255;
        float v;
        if (j < DH) v = h[(size_t)n * DH + j];
        else {
            int cn = col[(size_t)n * KD + r];
            v = h[(size_t)cn * DH + (j - DH)] - h[(size_t)n * DH + (j - DH)];
        }
        cat[wv][r][j] = v;
    }
    __syncthreads();
    int r = lane & 15, hi = lane >> 4;
    // GEMM1: t = lrelu(cat[16x256] @ W1[256x128] + b1)
    for (int nc = 0; nc < 8; ++nc) {
        v8f c = {};
        for (int k4 = 0; k4 < 256; k4 += 4) {
            v2f a, b;
            a.x = cat[wv][r][k4 + 2 * hi];
            a.y = cat[wv][r][k4 + 2 * hi + 1];
            int cc = nc * 16 + r;
            b.x = W1[(size_t)(k4 + 2 * hi)     * DH + cc];
            b.y = W1[(size_t)(k4 + 2 * hi + 1) * DH + cc];
            c = __builtin_amdgcn_wmma_f32_16x16x4_f32(false, a, false, b, (short)0, c, false, false);
        }
        float bv = b1[nc * 16 + r];
#pragma unroll
        for (int j = 0; j < 8; ++j) tb[wv][j + 8 * hi][nc * 16 + r] = lrelu(c[j] + bv);
    }
    __syncthreads();
    // GEMM2: msg = t[16x128] @ W2[128x128] + b2, then mean over 16 rows
    for (int nc = 0; nc < 8; ++nc) {
        v8f c = {};
        for (int k4 = 0; k4 < 128; k4 += 4) {
            v2f a, b;
            a.x = tb[wv][r][k4 + 2 * hi];
            a.y = tb[wv][r][k4 + 2 * hi + 1];
            int cc = nc * 16 + r;
            b.x = W2[(size_t)(k4 + 2 * hi)     * DH + cc];
            b.y = W2[(size_t)(k4 + 2 * hi + 1) * DH + cc];
            c = __builtin_amdgcn_wmma_f32_16x16x4_f32(false, a, false, b, (short)0, c, false, false);
        }
        float bv = b2[nc * 16 + r];
        float local = 0.f;
#pragma unroll
        for (int j = 0; j < 8; ++j) local += c[j] + bv;
        float other = __shfl_xor(local, 16, 32);
        float tot = (local + other) * (1.0f / 16.0f);
        if (hi == 0) out[(size_t)n * DH + nc * 16 + r] = tot;
    }
}

// ---------------- pooling ----------------
__global__ void k_gate_dot(const float* t, const float* W2, const float* b2, float* gate) {
    int n = blockIdx.x * blockDim.x + threadIdx.x;
    if (n >= NNODES) return;
    float s = b2[0];
    for (int d = 0; d < DH; ++d) s += t[(size_t)n * DH + d] * W2[d];
    gate[n] = s;
}

__global__ void k_pool_softmax(const float* gate, float* w_n) {
    int b = blockIdx.x, tid = threadIdx.x;
    __shared__ float red[256];
    __shared__ float sm, sden;
    float m = -3.4e38f;
    for (int i = tid; i < NPGR; i += 256) m = fmaxf(m, gate[b * NPGR + i]);
    red[tid] = m; __syncthreads();
    for (int o = 128; o > 0; o >>= 1) { if (tid < o) red[tid] = fmaxf(red[tid], red[tid + o]); __syncthreads(); }
    if (tid == 0) sm = red[0];
    __syncthreads();
    float s = 0.f;
    for (int i = tid; i < NPGR; i += 256) s += expf(gate[b * NPGR + i] - sm);
    red[tid] = s; __syncthreads();
    for (int o = 128; o > 0; o >>= 1) { if (tid < o) red[tid] += red[tid + o]; __syncthreads(); }
    if (tid == 0) sden = fmaxf(red[0], 1e-16f);
    __syncthreads();
    for (int i = tid; i < NPGR; i += 256)
        w_n[b * NPGR + i] = expf(gate[b * NPGR + i] - sm) / sden;
}

__global__ void k_pool_wsum(const float* h, const float* w_n, float* g) {
    int b = blockIdx.x, d = blockIdx.y, tid = threadIdx.x;
    __shared__ float red[256];
    float s = 0.f;
    for (int i = tid; i < NPGR; i += 256) {
        int n = b * NPGR + i;
        s += w_n[n] * h[(size_t)n * DH + d];
    }
    red[tid] = s; __syncthreads();
    for (int o = 128; o > 0; o >>= 1) { if (tid < o) red[tid] += red[tid + o]; __syncthreads(); }
    if (tid == 0) g[b * DH + d] = red[0];
}

// ---------------- final head (tiny MLP on [4,128]) ----------------
__global__ void k_head(const float* gp, const float* W1, const float* b1,
                       const float* lng, const float* lnb,
                       const float* W2, const float* b2,
                       const float* pW, const float* pb, float* out) {
    int d = threadIdx.x;   // 128 threads
    __shared__ float sg[128], y[128], red[128];
    __shared__ float mu, istd;
    for (int b = 0; b < NBAT; ++b) {
        sg[d] = gp[b * DH + d];
        __syncthreads();
        float a1 = b1[d];
        for (int k = 0; k < DH; ++k) a1 += sg[k] * W1[k * DH + d];
        a1 = lrelu(a1);
        __syncthreads();
        red[d] = a1; __syncthreads();
        for (int o = 64; o > 0; o >>= 1) { if (d < o) red[d] += red[d + o]; __syncthreads(); }
        if (d == 0) mu = red[0] / (float)DH;
        __syncthreads();
        float dv = a1 - mu;
        red[d] = dv * dv; __syncthreads();
        for (int o = 64; o > 0; o >>= 1) { if (d < o) red[d] += red[d + o]; __syncthreads(); }
        if (d == 0) istd = rsqrtf(red[0] / (float)DH + EPSB);
        __syncthreads();
        sg[d] = (a1 - mu) * istd * lng[d] + lnb[d];
        __syncthreads();
        float a2 = b2[d];
        for (int k = 0; k < DH; ++k) a2 += sg[k] * W2[k * DH + d];
        a2 = lrelu(a2);
        y[d] = a2;
        __syncthreads();
        if (d < KPCA) {
            float o = pb[d];
            for (int k = 0; k < DH; ++k) o += y[k] * pW[k * KPCA + d];
            out[b * KPCA + d] = o;
        }
        __syncthreads();
    }
}

// ---------------- host orchestration ----------------
extern "C" void kernel_launch(void* const* d_in, const int* in_sizes, int n_in,
                              void* d_out, int out_size, void* d_ws, size_t ws_size,
                              hipStream_t stream) {
    const float* x       = (const float*)d_in[0];
    const int*   ei      = (const int*)  d_in[1];
    const float* ea      = (const float*)d_in[2];
    const float* g1_Wl   = (const float*)d_in[4];
    const float* g1_bl   = (const float*)d_in[5];
    const float* g1_Wr   = (const float*)d_in[6];
    const float* g1_br   = (const float*)d_in[7];
    const float* g1_We   = (const float*)d_in[8];
    const float* g1_att  = (const float*)d_in[9];
    const float* g1_bias = (const float*)d_in[10];
    const float* bn1_g   = (const float*)d_in[11];
    const float* bn1_b   = (const float*)d_in[12];
    const float* em_W1   = (const float*)d_in[13];
    const float* em_b1   = (const float*)d_in[14];
    const float* em_W2   = (const float*)d_in[15];
    const float* em_b2   = (const float*)d_in[16];
    const float* bn2_g   = (const float*)d_in[17];
    const float* bn2_b   = (const float*)d_in[18];
    const float* g2_Wl   = (const float*)d_in[19];
    const float* g2_bl   = (const float*)d_in[20];
    const float* g2_Wr   = (const float*)d_in[21];
    const float* g2_br   = (const float*)d_in[22];
    const float* g2_att  = (const float*)d_in[23];
    const float* g2_bias = (const float*)d_in[24];
    const float* bn3_g   = (const float*)d_in[25];
    const float* bn3_b   = (const float*)d_in[26];
    const float* gate_W1 = (const float*)d_in[27];
    const float* gate_b1 = (const float*)d_in[28];
    const float* gate_W2 = (const float*)d_in[29];
    const float* gate_b2 = (const float*)d_in[30];
    const float* se_W1   = (const float*)d_in[31];
    const float* se_b1   = (const float*)d_in[32];
    const float* ln_g    = (const float*)d_in[33];
    const float* ln_b    = (const float*)d_in[34];
    const float* se_W2   = (const float*)d_in[35];
    const float* se_b2   = (const float*)d_in[36];
    const float* pca_W   = (const float*)d_in[37];
    const float* pca_b   = (const float*)d_in[38];
    float* out = (float*)d_out;

    char* wp = (char*)d_ws;
    auto alloc = [&](size_t bytes) -> void* {
        void* r = (void*)wp;
        wp += (bytes + 255) & ~(size_t)255;
        return r;
    };
    float* xl      = (float*)alloc((size_t)NNODES * 256 * 4);
    float* xr      = (float*)alloc((size_t)NNODES * 256 * 4);
    float* accb    = (float*)alloc((size_t)NNODES * 256 * 4);
    float* h1      = (float*)alloc((size_t)NNODES * DH * 4);
    float* h2      = (float*)alloc((size_t)NNODES * DH * 4);
    float* h3      = (float*)alloc((size_t)NNODES * DH * 4);
    float* tg      = (float*)alloc((size_t)NNODES * DH * 4);
    float* gram    = (float*)alloc((size_t)NPGR * NPGR * 4);
    float* ealoop  = (float*)alloc((size_t)NNODES * 6 * 4);
    float* cnt     = (float*)alloc((size_t)NNODES * 4);
    float* s_e     = (float*)alloc((size_t)NE1 * NHEAD * 4);
    float* m_nh    = (float*)alloc((size_t)NNODES * NHEAD * 4);
    float* den_nh  = (float*)alloc((size_t)NNODES * NHEAD * 4);
    float* bnstat  = (float*)alloc((size_t)DH * 2 * 4);
    float* sq      = (float*)alloc((size_t)NNODES * 4);
    float* gate    = (float*)alloc((size_t)NNODES * 4);
    float* w_n     = (float*)alloc((size_t)NNODES * 4);
    float* g_pool  = (float*)alloc((size_t)NBAT * DH * 4);
    int*   col     = (int*)  alloc((size_t)NNODES * KD * 4);
    __bf16* hb     = (__bf16*)alloc((size_t)NNODES * DH * 2);

    auto cdiv = [](int a, int b) { return (a + b - 1) / b; };
    const int EH = NE1 * NHEAD;

    // self-loop edge-attr mean
    k_fill<<<cdiv(NNODES, 256), 256, 0, stream>>>(cnt, 0.f, NNODES);
    k_fill<<<cdiv(NNODES * 6, 256), 256, 0, stream>>>(ealoop, 0.f, NNODES * 6);
    k_selfloop_sum<<<cdiv(NEDGES, 256), 256, 0, stream>>>(ei, ea, ealoop, cnt);
    k_selfloop_div<<<cdiv(NNODES * 6, 256), 256, 0, stream>>>(ealoop, cnt);

    // ---- GATv2 layer 1 ----
    k_lin6<<<cdiv(NNODES * 256, 256), 256, 0, stream>>>(x, g1_Wl, g1_bl, xl);
    k_lin6<<<cdiv(NNODES * 256, 256), 256, 0, stream>>>(x, g1_Wr, g1_br, xr);
    k_fill<<<cdiv(NNODES * NHEAD, 256), 256, 0, stream>>>(m_nh, -3.4e38f, NNODES * NHEAD);
    k_fill<<<cdiv(NNODES * NHEAD, 256), 256, 0, stream>>>(den_nh, 0.f, NNODES * NHEAD);
    k_fill<<<cdiv(NNODES * 256, 256), 256, 0, stream>>>(accb, 0.f, NNODES * 256);
    k_gat_score<<<cdiv(EH, 256), 256, 0, stream>>>(ei, xl, xr, ea, ealoop, g1_We, g1_att, s_e, m_nh);
    k_gat_exp<<<cdiv(EH, 256), 256, 0, stream>>>(ei, s_e, m_nh, den_nh);
    k_gat_accum<<<cdiv(EH, 256), 256, 0, stream>>>(ei, s_e, den_nh, xl, accb);
    k_headmean_bias<<<cdiv(NNODES * DH, 256), 256, 0, stream>>>(accb, g1_bias, h1);
    k_bn_stats<<<DH, 256, 0, stream>>>(h1, bnstat);
    k_bn_apply<<<cdiv(NNODES * DH, 256), 256, 0, stream>>>(h1, bnstat, bn1_g, bn1_b, 1);

    // ---- dynamic KNN (bf16 WMMA Gram + top-16) ----
    k_sqnorm<<<cdiv(NNODES, 256), 256, 0, stream>>>(h1, sq);
    k_cvt_bf16<<<cdiv(NNODES * DH, 256), 256, 0, stream>>>(h1, hb);
    for (int b = 0; b < NBAT; ++b) {
        k_gram<<<(NPGR / 16) * (NPGR / 16) / 8, 256, 0, stream>>>(hb, gram, b);
        k_topk<<<cdiv(NPGR, 256), 256, 0, stream>>>(gram, sq, col, b);
    }

    // ---- fused edge-MLP conv (fp32 WMMA) ----
    k_edge_mlp<<<NNODES / EMW, 32 * EMW, 0, stream>>>(h1, col, em_W1, em_b1, em_W2, em_b2, h2);
    k_bn_stats<<<DH, 256, 0, stream>>>(h2, bnstat);
    k_bn_apply<<<cdiv(NNODES * DH, 256), 256, 0, stream>>>(h2, bnstat, bn2_g, bn2_b, 1);

    // ---- GATv2 layer 2 (fp32 WMMA transforms) ----
    {
        int ntiles = (NNODES / 16) * (256 / 16);
        k_gemm_f32<<<cdiv(ntiles, 8), 256, 0, stream>>>(h2, g2_Wl, g2_bl, xl, NNODES, 256, DH, 0);
        k_gemm_f32<<<cdiv(ntiles, 8), 256, 0, stream>>>(h2, g2_Wr, g2_br, xr, NNODES, 256, DH, 0);
    }
    k_fill<<<cdiv(NNODES * NHEAD, 256), 256, 0, stream>>>(m_nh, -3.4e38f, NNODES * NHEAD);
    k_fill<<<cdiv(NNODES * NHEAD, 256), 256, 0, stream>>>(den_nh, 0.f, NNODES * NHEAD);
    k_fill<<<cdiv(NNODES * 256, 256), 256, 0, stream>>>(accb, 0.f, NNODES * 256);
    k_gat_score<<<cdiv(EH, 256), 256, 0, stream>>>(ei, xl, xr, nullptr, nullptr, nullptr, g2_att, s_e, m_nh);
    k_gat_exp<<<cdiv(EH, 256), 256, 0, stream>>>(ei, s_e, m_nh, den_nh);
    k_gat_accum<<<cdiv(EH, 256), 256, 0, stream>>>(ei, s_e, den_nh, xl, accb);
    k_headmean_bias<<<cdiv(NNODES * DH, 256), 256, 0, stream>>>(accb, g2_bias, h3);
    k_bn_stats<<<DH, 256, 0, stream>>>(h3, bnstat);
    k_bn_apply<<<cdiv(NNODES * DH, 256), 256, 0, stream>>>(h3, bnstat, bn3_g, bn3_b, 1);

    // ---- attentional pooling ----
    {
        int ntiles = (NNODES / 16) * (DH / 16);
        k_gemm_f32<<<cdiv(ntiles, 8), 256, 0, stream>>>(h3, gate_W1, gate_b1, tg, NNODES, DH, DH, 2);
    }
    k_gate_dot<<<cdiv(NNODES, 256), 256, 0, stream>>>(tg, gate_W2, gate_b2, gate);
    k_pool_softmax<<<NBAT, 256, 0, stream>>>(gate, w_n);
    k_pool_wsum<<<dim3(NBAT, DH), 256, 0, stream>>>(h3, w_n, g_pool);

    // ---- shape embed + PCA head ----
    k_head<<<1, DH, 0, stream>>>(g_pool, se_W1, se_b1, ln_g, ln_b, se_W2, se_b2, pca_W, pca_b, out);
    (void)in_sizes; (void)n_in; (void)out_size; (void)ws_size;
}
